// EncoderTreeRNN_42013370089577
// MI455X (gfx1250) — matrix operand order repaired and checked by reading, not compiled, not run on hardware
//
#include <hip/hip_runtime.h>
#include <hip/hip_bf16.h>

// ---------------------------------------------------------------------------
// Tree-GRU encoder, CDNA5 (gfx1250): v_wmma_f32_16x16x32_f16 matrix core +
// TDM (tensor_load_to_lds) weight-tile feeds with TENSORcnt synchronization.
// ---------------------------------------------------------------------------

typedef _Float16 v8h  __attribute__((ext_vector_type(8)));
typedef _Float16 v16h __attribute__((ext_vector_type(16)));
typedef float    v8f  __attribute__((ext_vector_type(8)));
typedef unsigned int u32x4 __attribute__((ext_vector_type(4)));
typedef int          i32x4 __attribute__((ext_vector_type(4)));
typedef int          i32x8 __attribute__((ext_vector_type(8)));

constexpr int cB = 64;
constexpr int cL = 512;
constexpr int cE = 512;
constexpr int cH = 1024;

constexpr int LDK = 40;   // 32 K-halves + 8 pad  (80B rows; pad = 4 DWORDs per 16 DWORDs)
constexpr int LDN = 72;   // 64 N-halves + 8 pad  (144B rows; pad = 4 DWORDs per 32 DWORDs)

__device__ __forceinline__ float sigm(float x) { return 1.0f / (1.0f + __expf(-x)); }

// ---------------------------------------------------------------------------
// TDM: 2D tile load, global -> LDS, with LDS row padding (ISA 08 §8).
// data_size = 2 bytes. Issued by one wave; tracked via TENSORcnt.
// ---------------------------------------------------------------------------
__device__ __forceinline__ void tdm_load_2d(unsigned lds_off, const void* gptr,
                                            unsigned tensor_d0, unsigned tensor_d1,
                                            unsigned d0_stride,
                                            unsigned tile_d0, unsigned tile_d1,
                                            unsigned pad_interval_code,
                                            unsigned pad_amount_code) {
  unsigned long long ga = (unsigned long long)(size_t)gptr;
  u32x4 g0;
  g0[0] = 1u;                                         // count=1, user descriptor
  g0[1] = lds_off;                                    // lds_addr (bytes)
  g0[2] = (unsigned)(ga & 0xffffffffu);               // global_addr[31:0]
  g0[3] = (unsigned)((ga >> 32) & 0x01ffffffu)        // global_addr[56:32]
        | (2u << 30);                                 // type = 2 (image)
  i32x8 g1;
  g1[0] = (int)((1u << 16)                            // data_size = 1 -> 2 bytes
        | (1u << 20)                                  // pad_enable
        | (pad_interval_code << 22)
        | (pad_amount_code << 25));
  g1[1] = (int)((tensor_d0 & 0xffffu) << 16);                                   // dim0[15:0]
  g1[2] = (int)(((tensor_d0 >> 16) & 0xffffu) | ((tensor_d1 & 0xffffu) << 16)); // dim0[31:16]|dim1[15:0]
  g1[3] = (int)(((tensor_d1 >> 16) & 0xffffu) | ((tile_d0 & 0xffffu) << 16));   // dim1[31:16]|tile0
  g1[4] = (int)(tile_d1 & 0xffffu);                   // tile_dim1 (tile_dim2 = 0)
  g1[5] = (int)d0_stride;                             // tensor_dim0_stride[31:0]
  g1[6] = 0;
  g1[7] = 0;
  i32x4 z4 = {0, 0, 0, 0};
#if defined(__clang_major__) && (__clang_major__ >= 23)
  i32x8 z8 = {0, 0, 0, 0, 0, 0, 0, 0};
  __builtin_amdgcn_tensor_load_to_lds(g0, g1, z4, z4, z8, 0);
#else
  __builtin_amdgcn_tensor_load_to_lds(g0, g1, z4, z4, 0);
#endif
}

__device__ __forceinline__ unsigned lds_off_u32(const void* p) {
  return (unsigned)(size_t)p;   // LDS aperture: addr[31:0] = LDS byte offset
}

// A-fragment (16 rows x 32 K, f16). ISA 7.12.2 16-bit A layout.
__device__ __forceinline__ v16h fragA(const _Float16* tile, int rowBase) {
  const int lane = threadIdx.x & 31;
  const _Float16* p = tile + (rowBase + (lane & 15)) * LDK + ((lane >> 4) << 3);
  v8h lo = *(const v8h*)(p);
  v8h hi = *(const v8h*)(p + 16);
  return __builtin_shufflevector(lo, hi, 0,1,2,3,4,5,6,7,8,9,10,11,12,13,14,15);
}

// B-fragment (32 K x 16 N, f16). Lane l holds column l%16; K in table order (8h+s, 16+8h+s).
__device__ __forceinline__ v16h fragB(const _Float16* tile, int colBase) {
  const int lane = threadIdx.x & 31;
  const _Float16* p = tile + ((lane >> 4) << 3) * LDN + colBase + (lane & 15);
  v16h r;
#pragma unroll
  for (int s = 0; s < 8; ++s) { r[s] = p[s * LDN]; r[s + 8] = p[(16 + s) * LDN]; }
  return r;
}

__device__ __forceinline__ v8f wmma_f16(v16h a, v16h b, v8f c) {
  return __builtin_amdgcn_wmma_f32_16x16x32_f16(false, a, false, b, (short)0, c, false, false);
}

// Load all 4 B-fragments of one 32x64 weight tile, then issue the 4 WMMAs —
// batches the strided 16-bit LDS loads so they overlap instead of a dscnt-0 drain per MMA.
__device__ __forceinline__ void gate4(const _Float16* sWt, v16h a, v8f* acc) {
  v16h b0 = fragB(sWt, 0);
  v16h b1 = fragB(sWt, 16);
  v16h b2 = fragB(sWt, 32);
  v16h b3 = fragB(sWt, 48);
  acc[0] = wmma_f16(a, b0, acc[0]);
  acc[1] = wmma_f16(a, b1, acc[1]);
  acc[2] = wmma_f16(a, b2, acc[2]);
  acc[3] = wmma_f16(a, b3, acc[3]);
}

// ---------------------------------------------------------------------------
__global__ __launch_bounds__(256)
void k_f32_to_f16(const float* __restrict__ src, _Float16* __restrict__ dst, int n) {
  int i = (blockIdx.x * 256 + threadIdx.x) * 4;
  if (i + 3 < n) {
    float4 f = *(const float4*)(src + i);
    dst[i + 0] = (_Float16)f.x;
    dst[i + 1] = (_Float16)f.y;
    dst[i + 2] = (_Float16)f.z;
    dst[i + 3] = (_Float16)f.w;
  }
}

__global__ __launch_bounds__(256)
void k_bias(const float* b_z, const float* b_zl, const float* b_zr,
            const float* b_r, const float* b_rl, const float* b_rr,
            const float* b_h, const float* b_hl, const float* b_hr,
            float* bz, float* br, float* bh) {
  int i = blockIdx.x * 256 + threadIdx.x;
  if (i < cH) {
    bz[i] = b_z[i] + b_zl[i] + b_zr[i];
    br[i] = b_r[i] + b_rl[i] + b_rr[i];
    bh[i] = b_h[i] + b_hl[i] + b_hr[i];
  }
}

// ---------------------------------------------------------------------------
// Level 0: h = (1 - sigmoid(word@Wz + bz)) * tanh(word@Wh + bh)
// A = emb[tokens] gather (manual, fp32->fp16 in flight); W tiles via TDM.
// ---------------------------------------------------------------------------
__global__ __launch_bounds__(256)
void k_level0(const int* __restrict__ tokens, const float* __restrict__ emb,
              const _Float16* __restrict__ Wz, const _Float16* __restrict__ Wh,
              const float* __restrict__ bz, const float* __restrict__ bh,
              _Float16* __restrict__ hout) {
  __shared__ _Float16 sA[128 * LDK];
  __shared__ _Float16 sWz[32 * LDN];
  __shared__ _Float16 sWh[32 * LDN];
  const int tid = threadIdx.x, lane = tid & 31, wave = tid >> 5;
  const int l15 = lane & 15;
  const int n0 = blockIdx.x * 64, m0 = blockIdx.y * 128;

  v8f accZ[4], accH[4];
#pragma unroll
  for (int t = 0; t < 4; ++t) {
    float z0 = bz[n0 + t * 16 + l15];
    float h0 = bh[n0 + t * 16 + l15];
#pragma unroll
    for (int v = 0; v < 8; ++v) { accZ[t][v] = z0; accH[t][v] = h0; }
  }

  for (int kb = 0; kb < cE; kb += 32) {
    if (wave == 0) {                       // one wave drives the TDM; EXEC ignored
      tdm_load_2d(lds_off_u32(sWz), Wz + (size_t)kb * cH + n0,
                  cH - n0, cE - kb, cH, 64, 32, 4, 3);
      tdm_load_2d(lds_off_u32(sWh), Wh + (size_t)kb * cH + n0,
                  cH - n0, cE - kb, cH, 64, 32, 4, 3);
    }
    // A tile: 128 rows x 32 halves, gathered + converted from fp32 emb
#pragma unroll
    for (int i = 0; i < 2; ++i) {
      int c = tid + i * 256;
      int row = c >> 2, seg = c & 3;
      int tok = tokens[m0 + row];
      const float4* src = (const float4*)(emb + (size_t)tok * cE + kb + seg * 8);
      float4 f0 = src[0], f1 = src[1];
      _Float16* d = sA + row * LDK + seg * 8;
      d[0] = (_Float16)f0.x; d[1] = (_Float16)f0.y; d[2] = (_Float16)f0.z; d[3] = (_Float16)f0.w;
      d[4] = (_Float16)f1.x; d[5] = (_Float16)f1.y; d[6] = (_Float16)f1.z; d[7] = (_Float16)f1.w;
    }
    if (wave == 0) __builtin_amdgcn_s_wait_tensorcnt(0);
    __syncthreads();
    v16h a = fragA(sA, wave * 16);
    gate4(sWz, a, accZ);
    gate4(sWh, a, accH);
    __syncthreads();
  }

#pragma unroll
  for (int v = 0; v < 8; ++v) {
    int row = m0 + wave * 16 + ((lane >> 4) << 3) + v;
    size_t base = (size_t)row * cH;
#pragma unroll
    for (int t = 0; t < 4; ++t) {
      int col = n0 + t * 16 + l15;
      float z = sigm(accZ[t][v]);
      float ht = tanhf(accH[t][v]);
      hout[base + col] = (_Float16)((1.0f - z) * ht);
    }
  }
}

// ---------------------------------------------------------------------------
// Phase 1: z = sigmoid(hl@Uzl + hr@Uzr + bz); r = sigmoid(hl@Url + hr@Urr + br)
// emits zbuf = z, rhl = r*hl, rhr = r*hr. A (stride-2 child gather) manual,
// 4 U tiles via TDM.
// ---------------------------------------------------------------------------
__global__ __launch_bounds__(256)
void k_phase1(const _Float16* __restrict__ hin, int half, int rowsOut,
              const _Float16* __restrict__ Uzl, const _Float16* __restrict__ Uzr,
              const _Float16* __restrict__ Url, const _Float16* __restrict__ Urr,
              const float* __restrict__ bz, const float* __restrict__ br,
              _Float16* __restrict__ zb, _Float16* __restrict__ rhl,
              _Float16* __restrict__ rhr) {
  __shared__ _Float16 sHL[128 * LDK];
  __shared__ _Float16 sHR[128 * LDK];
  __shared__ _Float16 sW[4][32 * LDN];
  const int tid = threadIdx.x, lane = tid & 31, wave = tid >> 5;
  const int l15 = lane & 15;
  const int n0 = blockIdx.x * 64, m0 = blockIdx.y * 128;

  v8f accZ[4], accR[4];
#pragma unroll
  for (int t = 0; t < 4; ++t) {
    float z0 = bz[n0 + t * 16 + l15];
    float r0 = br[n0 + t * 16 + l15];
#pragma unroll
    for (int v = 0; v < 8; ++v) { accZ[t][v] = z0; accR[t][v] = r0; }
  }

  for (int kb = 0; kb < cH; kb += 32) {
    if (wave == 0) {
      tdm_load_2d(lds_off_u32(sW[0]), Uzl + (size_t)kb * cH + n0, cH - n0, cH - kb, cH, 64, 32, 4, 3);
      tdm_load_2d(lds_off_u32(sW[1]), Uzr + (size_t)kb * cH + n0, cH - n0, cH - kb, cH, 64, 32, 4, 3);
      tdm_load_2d(lds_off_u32(sW[2]), Url + (size_t)kb * cH + n0, cH - n0, cH - kb, cH, 64, 32, 4, 3);
      tdm_load_2d(lds_off_u32(sW[3]), Urr + (size_t)kb * cH + n0, cH - n0, cH - kb, cH, 64, 32, 4, 3);
    }
#pragma unroll
    for (int i = 0; i < 2; ++i) {
      int c = tid + i * 256;
      int row = c >> 2, seg = c & 3;
      int ro = m0 + row;
      v8h zl = {}, zr = {};
      if (ro < rowsOut) {
        int b = ro / half, ii = ro - b * half;
        size_t hlrow = (size_t)(b * (2 * half) + 2 * ii);
        const _Float16* p = hin + hlrow * cH + kb + seg * 8;
        zl = *(const v8h*)(p);
        zr = *(const v8h*)(p + cH);
      }
      *(v8h*)(sHL + row * LDK + seg * 8) = zl;
      *(v8h*)(sHR + row * LDK + seg * 8) = zr;
    }
    if (wave == 0) __builtin_amdgcn_s_wait_tensorcnt(0);
    __syncthreads();
    v16h al = fragA(sHL, wave * 16);
    v16h ar = fragA(sHR, wave * 16);
    gate4(sW[0], al, accZ);
    gate4(sW[1], ar, accZ);
    gate4(sW[2], al, accR);
    gate4(sW[3], ar, accR);
    __syncthreads();
  }

#pragma unroll
  for (int v = 0; v < 8; ++v) {
    int ro = m0 + wave * 16 + ((lane >> 4) << 3) + v;
    if (ro >= rowsOut) continue;
    int b = ro / half, ii = ro - b * half;
    size_t hlrow = (size_t)(b * (2 * half) + 2 * ii);
    size_t obase = (size_t)ro * cH;
#pragma unroll
    for (int t = 0; t < 4; ++t) {
      int col = n0 + t * 16 + l15;
      float z = sigm(accZ[t][v]);
      float r = sigm(accR[t][v]);
      float hl = (float)hin[hlrow * cH + col];
      float hr = (float)hin[(hlrow + 1) * cH + col];
      zb[obase + col]  = (_Float16)z;
      rhl[obase + col] = (_Float16)(r * hl);
      rhr[obase + col] = (_Float16)(r * hr);
    }
  }
}

// ---------------------------------------------------------------------------
// Phase 2: h = z*(hl+hr) + (1-z)*tanh(rhl@Uhl + rhr@Uhr + bh)
// A tiles (rhl/rhr, contiguous rows) AND U tiles all via TDM.
// ---------------------------------------------------------------------------
__global__ __launch_bounds__(256)
void k_phase2(const _Float16* __restrict__ hin, int half, int rowsOut,
              const _Float16* __restrict__ rhl, const _Float16* __restrict__ rhr,
              const _Float16* __restrict__ Uhl, const _Float16* __restrict__ Uhr,
              const float* __restrict__ bh, const _Float16* __restrict__ zb,
              _Float16* __restrict__ hout) {
  __shared__ _Float16 sA1[128 * LDK];
  __shared__ _Float16 sA2[128 * LDK];
  __shared__ _Float16 sW[2][32 * LDN];
  const int tid = threadIdx.x, lane = tid & 31, wave = tid >> 5;
  const int l15 = lane & 15;
  const int n0 = blockIdx.x * 64, m0 = blockIdx.y * 128;
  const unsigned rowsRem = (unsigned)(rowsOut - m0);   // >=1; TDM zero-fills beyond

  v8f acc[4];
#pragma unroll
  for (int t = 0; t < 4; ++t) {
    float h0 = bh[n0 + t * 16 + l15];
#pragma unroll
    for (int v = 0; v < 8; ++v) acc[t][v] = h0;
  }

  for (int kb = 0; kb < cH; kb += 32) {
    if (wave == 0) {
      // A tiles: 128 rows x 32 halves, row stride cH, LDS stride 40 halves
      tdm_load_2d(lds_off_u32(sA1), rhl + (size_t)m0 * cH + kb, cH - kb, rowsRem, cH, 32, 128, 3, 3);
      tdm_load_2d(lds_off_u32(sA2), rhr + (size_t)m0 * cH + kb, cH - kb, rowsRem, cH, 32, 128, 3, 3);
      // weight tiles: 32 K-rows x 64 N
      tdm_load_2d(lds_off_u32(sW[0]), Uhl + (size_t)kb * cH + n0, cH - n0, cH - kb, cH, 64, 32, 4, 3);
      tdm_load_2d(lds_off_u32(sW[1]), Uhr + (size_t)kb * cH + n0, cH - n0, cH - kb, cH, 64, 32, 4, 3);
      __builtin_amdgcn_s_wait_tensorcnt(0);
    }
    __syncthreads();
    v16h a1 = fragA(sA1, wave * 16);
    v16h a2 = fragA(sA2, wave * 16);
    gate4(sW[0], a1, acc);
    gate4(sW[1], a2, acc);
    __syncthreads();
  }

#pragma unroll
  for (int v = 0; v < 8; ++v) {
    int ro = m0 + wave * 16 + ((lane >> 4) << 3) + v;
    if (ro >= rowsOut) continue;
    int b = ro / half, ii = ro - b * half;
    size_t hlrow = (size_t)(b * (2 * half) + 2 * ii);
    size_t obase = (size_t)ro * cH;
#pragma unroll
    for (int t = 0; t < 4; ++t) {
      int col = n0 + t * 16 + l15;
      float th = tanhf(acc[t][v]);
      float z  = (float)zb[obase + col];
      float hl = (float)hin[hlrow * cH + col];
      float hr = (float)hin[(hlrow + 1) * cH + col];
      hout[obase + col] = (_Float16)(z * (hl + hr) + (1.0f - z) * th);
    }
  }
}

__global__ __launch_bounds__(256)
void k_out(const _Float16* __restrict__ h, float* __restrict__ out, int n) {
  int i = blockIdx.x * 256 + threadIdx.x;
  if (i < n) out[i] = (float)h[i];
}

// ---------------------------------------------------------------------------
extern "C" void kernel_launch(void* const* d_in, const int* in_sizes, int n_in,
                              void* d_out, int out_size, void* d_ws, size_t ws_size,
                              hipStream_t stream) {
  const int*   tokens = (const int*)d_in[0];
  const float* emb    = (const float*)d_in[1];
  const float* W_z  = (const float*)d_in[2];
  const float* b_z  = (const float*)d_in[3];
  const float* U_zl = (const float*)d_in[4];
  const float* b_zl = (const float*)d_in[5];
  const float* U_zr = (const float*)d_in[6];
  const float* b_zr = (const float*)d_in[7];
  const float* W_r  = (const float*)d_in[8];
  const float* b_r  = (const float*)d_in[9];
  const float* U_rl = (const float*)d_in[10];
  const float* b_rl = (const float*)d_in[11];
  const float* U_rr = (const float*)d_in[12];
  const float* b_rr = (const float*)d_in[13];
  const float* W_h  = (const float*)d_in[14];
  const float* b_h  = (const float*)d_in[15];
  const float* U_hl = (const float*)d_in[16];
  const float* b_hl = (const float*)d_in[17];
  const float* U_hr = (const float*)d_in[18];
  const float* b_hr = (const float*)d_in[19];

  const size_t EH = (size_t)cE * cH;
  const size_t HH = (size_t)cH * cH;

  char* p = (char*)d_ws;
  auto take = [&](size_t bytes) {
    void* r = (void*)p;
    p += (bytes + 255) & ~(size_t)255;
    return r;
  };
  _Float16* Wz16  = (_Float16*)take(EH * 2);
  _Float16* Wr16  = (_Float16*)take(EH * 2);   // converted but unused at level 0 (r gate is moot there)
  _Float16* Wh16  = (_Float16*)take(EH * 2);
  _Float16* Uzl16 = (_Float16*)take(HH * 2);
  _Float16* Uzr16 = (_Float16*)take(HH * 2);
  _Float16* Url16 = (_Float16*)take(HH * 2);
  _Float16* Urr16 = (_Float16*)take(HH * 2);
  _Float16* Uhl16 = (_Float16*)take(HH * 2);
  _Float16* Uhr16 = (_Float16*)take(HH * 2);
  float* bzc = (float*)take(cH * 4);
  float* brc = (float*)take(cH * 4);
  float* bhc = (float*)take(cH * 4);
  _Float16* bufA = (_Float16*)take((size_t)cB * cL       * cH * 2); // 64 MB
  _Float16* bufB = (_Float16*)take((size_t)cB * (cL / 2) * cH * 2); // 32 MB
  _Float16* zbuf = (_Float16*)take((size_t)cB * (cL / 2) * cH * 2);
  _Float16* rhlb = (_Float16*)take((size_t)cB * (cL / 2) * cH * 2);
  _Float16* rhrb = (_Float16*)take((size_t)cB * (cL / 2) * cH * 2);

  k_f32_to_f16<<<(int)(EH / 1024), 256, 0, stream>>>(W_z,  Wz16,  (int)EH);
  k_f32_to_f16<<<(int)(EH / 1024), 256, 0, stream>>>(W_r,  Wr16,  (int)EH);
  k_f32_to_f16<<<(int)(EH / 1024), 256, 0, stream>>>(W_h,  Wh16,  (int)EH);
  k_f32_to_f16<<<(int)(HH / 1024), 256, 0, stream>>>(U_zl, Uzl16, (int)HH);
  k_f32_to_f16<<<(int)(HH / 1024), 256, 0, stream>>>(U_zr, Uzr16, (int)HH);
  k_f32_to_f16<<<(int)(HH / 1024), 256, 0, stream>>>(U_rl, Url16, (int)HH);
  k_f32_to_f16<<<(int)(HH / 1024), 256, 0, stream>>>(U_rr, Urr16, (int)HH);
  k_f32_to_f16<<<(int)(HH / 1024), 256, 0, stream>>>(U_hl, Uhl16, (int)HH);
  k_f32_to_f16<<<(int)(HH / 1024), 256, 0, stream>>>(U_hr, Uhr16, (int)HH);
  k_bias<<<4, 256, 0, stream>>>(b_z, b_zl, b_zr, b_r, b_rl, b_rr, b_h, b_hl, b_hr,
                                bzc, brc, bhc);

  {
    dim3 grid(cH / 64, (cB * cL) / 128);
    k_level0<<<grid, 256, 0, stream>>>(tokens, emb, Wz16, Wh16, bzc, bhc, bufA);
  }

  _Float16* cur = bufA;
  _Float16* nxt = bufB;
  for (int n = cL; n > 1; n >>= 1) {
    int half = n >> 1;
    int rowsOut = cB * half;
    dim3 grid(cH / 64, (rowsOut + 127) / 128);
    k_phase1<<<grid, 256, 0, stream>>>(cur, half, rowsOut,
                                       Uzl16, Uzr16, Url16, Urr16,
                                       bzc, brc, zbuf, rhlb, rhrb);
    k_phase2<<<grid, 256, 0, stream>>>(cur, half, rowsOut,
                                       rhlb, rhrb, Uhl16, Uhr16,
                                       bhc, zbuf, nxt);
    _Float16* t = cur; cur = nxt; nxt = t;
  }

  k_out<<<(cB * cH) / 256, 256, 0, stream>>>(cur, (float*)d_out, cB * cH);
}
